// MixHop_81527069213106
// MI455X (gfx1250) — matrix operand order repaired and checked by reading, not compiled, not run on hardware
//
#include <hip/hip_runtime.h>

// ---------------------------------------------------------------------------
// CDNA5 (gfx1250, wave32) MixHop pipeline.
// GEMM operands pre-converted to bf16:
//   A operands: row-major bf16.
//   B operands: fragment-major packed-pair blocks:
//     dword[((kk>>4)*Nw + n)*16 + (kk&15)] = (B[2kk][n], B[2kk+1][n])
//     -> per column, 16 fragment dwords contiguous (64B) per 32-K block.
// Staging uses GLOBAL_LOAD_ASYNC_TO_LDS_B128 (no VGPR data, ASYNCcnt),
// double-buffered LDS, one s_wait_asynccnt + barrier per K-iteration.
// ---------------------------------------------------------------------------

typedef __bf16 v16bf __attribute__((ext_vector_type(16)));
typedef float  v8f   __attribute__((ext_vector_type(8)));

union FragU {
    unsigned int u[8];   // 8 dwords = 16 bf16
    v16bf        bf;
};

__device__ __forceinline__ unsigned short f2bf(float f) {
    unsigned int x = __builtin_bit_cast(unsigned int, f);
    x += 0x7FFFu + ((x >> 16) & 1u);          // round-to-nearest-even
    return (unsigned short)(x >> 16);
}

// memory(global) -> LDS, 16B per lane, tracked by ASYNCcnt (ISA 15.18.3 op 98)
__device__ __forceinline__ void async_g2l_b128(unsigned lds_byte_off, const void* gaddr) {
    asm volatile("global_load_async_to_lds_b128 %0, %1, off"
                 :: "v"(lds_byte_off),
                    "v"((unsigned long long)(size_t)gaddr)
                 : "memory");
}
__device__ __forceinline__ void wait_async0() {
    asm volatile("s_wait_asynccnt 0x0" ::: "memory");
}

// ---------------------------------------------------------------------------
// GEMM: D = A(bf16 rm) x B(bf16 fragment-major packed) + bias, 128x128 tile,
// K-step 64. 8 waves in 4(m) x 2(n); each wave: 2x4 WMMA 16x16 accumulators.
// omode: 0 = f32 row-major, 1 = bf16 row-major, 2 = bf16 fragment-major packed.
// ---------------------------------------------------------------------------
__global__ __launch_bounds__(256, 1) void gemm_bf16_wmma(
    const unsigned short* __restrict__ A, int lda,     // bf16 row-major [M][lda]
    const unsigned int*  __restrict__ Bp, int ldbp,    // fragment-major packed, Nw=ldbp
    void* __restrict__ Cout, int ldc, int omode,
    const float* __restrict__ bias,
    int M, int N, int K)
{
    // A tile: [row][k 0..63] bf16, row stride 72 (fragment dwords contiguous).
    __shared__ __align__(16) unsigned short As[2][128 * 72];
    // B tile, fragment-major: dword index (ks*128 + col)*20 + hh*8 + j.
    __shared__ __align__(16) unsigned int   Bs[2][2 * 128 * 20];

    const int tid  = threadIdx.x;
    const int lane = tid & 31;
    const int wave = tid >> 5;
    const int wm   = wave >> 1;        // 0..3
    const int wn   = wave & 1;         // 0..1
    const int row0 = blockIdx.y * 128;
    const int col0 = blockIdx.x * 128;
    const int hh   = lane >> 4;        // lane half
    const int ll   = lane & 15;

    // ---- hoisted staging addresses (independent of k-iteration) ----
    size_t aG[4]; unsigned aL[4];      // aL/bL: LDS byte offsets within buffer 0
    size_t bG[4]; unsigned bL[4];
    #pragma unroll
    for (int i = 0; i < 4; ++i) {
        int idx = tid + i * 256;                 // 0..1023
        int r   = idx >> 3;                      // 0..127
        int ko  = (idx & 7) * 8;                 // 0,8,..,56
        aG[i] = (size_t)(row0 + r) * lda + ko;
        aL[i] = (unsigned)(r * 72 + ko) * 2;
        int n  = idx & 127;                      // column in tile
        int rr = idx >> 7;                       // 0..7
        int jq = (rr & 1) * 4;                   // j quad
        int h2 = (rr >> 1) & 1;                  // lane-half slot
        int ks = rr >> 2;                        // k substep
        bG[i] = (size_t)ks * 16 * ldbp + (size_t)(col0 + n) * 16 + h2 * 8 + jq;
        bL[i] = (unsigned)((ks * 128 + n) * 20 + h2 * 8 + jq) * 4;
    }

    v8f acc[2][4];
    for (int i = 0; i < 2; ++i)
        for (int j = 0; j < 4; ++j)
            for (int r = 0; r < 8; ++r) acc[i][j][r] = 0.0f;

    auto issue = [&](int kt, int buf) {
        unsigned aBase = (unsigned)(size_t)&As[buf][0];
        unsigned bBase = (unsigned)(size_t)&Bs[buf][0];
        #pragma unroll
        for (int i = 0; i < 4; ++i)
            async_g2l_b128(aBase + aL[i], A + aG[i] + (size_t)kt * 64);
        #pragma unroll
        for (int i = 0; i < 4; ++i)
            async_g2l_b128(bBase + bL[i], Bp + bG[i] + (size_t)kt * 32 * ldbp);
    };

    // ---- pipeline prologue ----
    issue(0, 0);
    wait_async0();
    __syncthreads();

    const int KT = K >> 6;                       // K / 64
    for (int kt = 0; kt < KT; ++kt) {
        const int cur  = kt & 1;
        const bool more = (kt + 1 < KT);
        if (more) issue(kt + 1, cur ^ 1);        // async engine fills other buffer

        #pragma unroll
        for (int ks = 0; ks < 2; ++ks) {
            // A frags: lane ll = row M; vgpr j -> K = hh*16+2j,+1
            FragU afr[2];
            #pragma unroll
            for (int mt = 0; mt < 2; ++mt) {
                const unsigned short* ap =
                    &As[cur][(wm * 32 + mt * 16 + ll) * 72 + ks * 32];
                *(uint4*)&afr[mt].u[0] = *(const uint4*)(ap + hh * 16);
                *(uint4*)&afr[mt].u[4] = *(const uint4*)(ap + hh * 16 + 8);
            }
            // B frags: fragment-major LDS -> two contiguous b128 loads
            FragU bfr[4];
            #pragma unroll
            for (int nt = 0; nt < 4; ++nt) {
                const unsigned int* bp =
                    &Bs[cur][(ks * 128 + wn * 64 + nt * 16 + ll) * 20 + hh * 8];
                *(uint4*)&bfr[nt].u[0] = *(const uint4*)(bp);
                *(uint4*)&bfr[nt].u[4] = *(const uint4*)(bp + 4);
            }
            #pragma unroll
            for (int mt = 0; mt < 2; ++mt)
                #pragma unroll
                for (int nt = 0; nt < 4; ++nt)
                    acc[mt][nt] = __builtin_amdgcn_wmma_f32_16x16x32_bf16(
                        false, afr[mt].bf, false, bfr[nt].bf,
                        (short)0, acc[mt][nt], false, false);
        }

        if (more) wait_async0();                 // next tile fully in LDS
        __syncthreads();
    }

    // ---- epilogue: C/D layout — vgpr r -> M = 8*hh + r, N = ll ----
    #pragma unroll
    for (int nt = 0; nt < 4; ++nt) {
        int n = col0 + wn * 64 + nt * 16 + ll;
        float bv = bias[n];
        #pragma unroll
        for (int mt = 0; mt < 2; ++mt) {
            int mb = row0 + wm * 32 + mt * 16 + hh * 8;    // even, mb/2 multiple of 4
            if (omode == 0) {
                float* C = (float*)Cout;
                #pragma unroll
                for (int r = 0; r < 8; ++r)
                    C[(size_t)(mb + r) * ldc + n] = acc[mt][nt][r] + bv;
            } else if (omode == 1) {
                unsigned short* C = (unsigned short*)Cout;
                #pragma unroll
                for (int r = 0; r < 8; ++r)
                    C[(size_t)(mb + r) * ldc + n] = f2bf(acc[mt][nt][r] + bv);
            } else {
                // fragment-major packed output (ldc = Nw of this matrix)
                unsigned int* C = (unsigned int*)Cout;
                int pr0 = mb >> 1;                      // packed row, multiple of 4
                uint4 o;
                o.x = (unsigned int)f2bf(acc[mt][nt][0] + bv) |
                      ((unsigned int)f2bf(acc[mt][nt][1] + bv) << 16);
                o.y = (unsigned int)f2bf(acc[mt][nt][2] + bv) |
                      ((unsigned int)f2bf(acc[mt][nt][3] + bv) << 16);
                o.z = (unsigned int)f2bf(acc[mt][nt][4] + bv) |
                      ((unsigned int)f2bf(acc[mt][nt][5] + bv) << 16);
                o.w = (unsigned int)f2bf(acc[mt][nt][6] + bv) |
                      ((unsigned int)f2bf(acc[mt][nt][7] + bv) << 16);
                *(uint4*)&C[((size_t)(pr0 >> 4) * ldc + n) * 16 + (pr0 & 15)] = o;
            }
        }
    }
}

// ---------------------------------------------------------------------------
// Converters
// ---------------------------------------------------------------------------
__global__ void f32_to_bf16_vec(const float* __restrict__ src,
                                unsigned short* __restrict__ dst, long n4)
{
    long i = (long)blockIdx.x * blockDim.x + threadIdx.x;
    if (i >= n4) return;
    float4 v = ((const float4*)src)[i];
    uint2 o;
    o.x = (unsigned int)f2bf(v.x) | ((unsigned int)f2bf(v.y) << 16);
    o.y = (unsigned int)f2bf(v.z) | ((unsigned int)f2bf(v.w) << 16);
    ((uint2*)dst)[i] = o;
}

// f32 [K][Nw] -> fragment-major packed bf16 pairs
__global__ void f32_to_bf16_pk(const float* __restrict__ src, int ld,
                               unsigned int* __restrict__ dst, int K, int Nw)
{
    int idx = blockIdx.x * blockDim.x + threadIdx.x;
    int total = (K / 2) * Nw;
    if (idx >= total) return;
    int n = idx % Nw, kk = idx / Nw;
    unsigned int d = (unsigned int)f2bf(src[(size_t)(2 * kk) * ld + n]) |
                     ((unsigned int)f2bf(src[(size_t)(2 * kk + 1) * ld + n]) << 16);
    dst[((size_t)(kk >> 4) * Nw + n) * 16 + (kk & 15)] = d;
}

// ---------------------------------------------------------------------------
// BatchNorm (training-mode column stats) + ReLU -> bf16 row-major
// ---------------------------------------------------------------------------
__global__ void zero_f32(float* p, int n) {
    int i = blockIdx.x * blockDim.x + threadIdx.x;
    if (i < n) p[i] = 0.0f;
}

__global__ void col_partial_stats(const float* __restrict__ H, int ldh, int rowsPerBlock,
                                  float* __restrict__ sums, float* __restrict__ sqs)
{
    int c  = blockIdx.x * blockDim.x + threadIdx.x;
    int r0 = blockIdx.y * rowsPerBlock;
    float s = 0.f, ss = 0.f;
    for (int r = 0; r < rowsPerBlock; ++r) {
        float v = H[(size_t)(r0 + r) * ldh + c];
        s += v; ss += v * v;
    }
    atomicAdd(&sums[c], s);
    atomicAdd(&sqs[c], ss);
}

__global__ void bn_relu_to_bf16(const float* __restrict__ H, unsigned short* __restrict__ Hb,
                                int total, int cols, float invM,
                                const float* __restrict__ sums, const float* __restrict__ sqs,
                                const float* __restrict__ gamma, const float* __restrict__ beta)
{
    int i = blockIdx.x * blockDim.x + threadIdx.x;
    if (i >= total) return;
    int c = i % cols;
    float mu  = sums[c] * invM;
    float var = sqs[c] * invM - mu * mu;
    float v = (H[i] - mu) * rsqrtf(var + 1e-5f) * gamma[c] + beta[c];
    Hb[i] = f2bf(v > 0.f ? v : 0.f);
}

// ---------------------------------------------------------------------------
// Launcher
// ---------------------------------------------------------------------------
extern "C" void kernel_launch(void* const* d_in, const int* in_sizes, int n_in,
                              void* d_out, int out_size, void* d_ws, size_t ws_size,
                              hipStream_t stream)
{
    (void)in_sizes; (void)n_in; (void)out_size; (void)ws_size;
    const int N = 8192, IN = 128, H = 512, OUT = 256;

    const float* x     = (const float*)d_in[0];
    const float* adj   = (const float*)d_in[1];
    const float* W1    = (const float*)d_in[2];
    const float* b1    = (const float*)d_in[3];
    const float* W2    = (const float*)d_in[4];
    const float* b2    = (const float*)d_in[5];
    const float* gamma = (const float*)d_in[6];
    const float* beta  = (const float*)d_in[7];
    const float* Wf    = (const float*)d_in[8];
    const float* bf    = (const float*)d_in[9];
    float* out = (float*)d_out;

    // ---- workspace layout (256B-aligned chunks) ----
    char* base = (char*)d_ws;
    size_t off = 0;
    auto alloc = [&](size_t bytes) -> void* {
        void* p = base + off;
        off = (off + bytes + 255) & ~(size_t)255;
        return p;
    };
    unsigned short* adj_b = (unsigned short*)alloc((size_t)N * N * 2);     // 128MB
    unsigned short* x_b   = (unsigned short*)alloc((size_t)N * IN * 2);
    unsigned int*   W1p   = (unsigned int*)alloc((size_t)3 * (IN / 2) * H * 4);
    unsigned int*   W2p   = (unsigned int*)alloc((size_t)3 * (1536 / 2) * OUT * 4);
    unsigned int*   Wfp   = (unsigned int*)alloc((size_t)(768 / 2) * OUT * 4);
    float*          H1    = (float*)alloc((size_t)N * 1536 * 4);           // pre-BN f32
    unsigned short* H1b   = (unsigned short*)alloc((size_t)N * 1536 * 2);  // post BN+ReLU
    unsigned short* H2b   = (unsigned short*)alloc((size_t)N * 768 * 2);
    unsigned int*   T1p   = (unsigned int*)alloc((size_t)(N / 2) * H * 4);
    unsigned int*   T2p   = (unsigned int*)alloc((size_t)(N / 2) * H * 4);
    unsigned int*   U1p   = (unsigned int*)alloc((size_t)(N / 2) * OUT * 4);
    unsigned int*   U2p   = (unsigned int*)alloc((size_t)(N / 2) * OUT * 4);
    unsigned int*   Vp    = (unsigned int*)alloc((size_t)(N / 2) * OUT * 4);
    float*          sums  = (float*)alloc((size_t)3 * 1536 * 4);           // sums|sqs|zbias
    float*          sqs   = sums + 1536;
    float*          zbias = sums + 3072;

    auto gemm = [&](const unsigned short* A, int lda, const unsigned int* Bp, int ldbp,
                    void* C, int ldc, int omode, const float* bias, int M, int Nn, int K) {
        dim3 grid(Nn / 128, M / 128, 1);
        gemm_bf16_wmma<<<grid, 256, 0, stream>>>(A, lda, Bp, ldbp, C, ldc, omode, bias, M, Nn, K);
    };

    // ---- one-time operand conversion (dominated by adj: 256MB read, 128MB write) ----
    {
        long n4 = (long)N * N / 4;
        f32_to_bf16_vec<<<dim3((unsigned)((n4 + 255) / 256)), 256, 0, stream>>>(adj, adj_b, n4);
        long x4 = (long)N * IN / 4;
        f32_to_bf16_vec<<<dim3((unsigned)((x4 + 255) / 256)), 256, 0, stream>>>(x, x_b, x4);
        for (int j = 0; j < 3; ++j)
            f32_to_bf16_pk<<<dim3(((IN / 2) * H + 255) / 256), 256, 0, stream>>>(
                W1 + (size_t)j * IN * H, H, W1p + (size_t)j * (IN / 2) * H, IN, H);
        for (int j = 0; j < 3; ++j)
            f32_to_bf16_pk<<<dim3(((1536 / 2) * OUT + 255) / 256), 256, 0, stream>>>(
                W2 + (size_t)j * 1536 * OUT, OUT, W2p + (size_t)j * (1536 / 2) * OUT, 1536, OUT);
        f32_to_bf16_pk<<<dim3(((768 / 2) * OUT + 255) / 256), 256, 0, stream>>>(
            Wf, OUT, Wfp, 768, OUT);
        zero_f32<<<dim3((3 * 1536 + 255) / 256), 256, 0, stream>>>(sums, 3 * 1536);
    }

    // ---- Layer 1 ----
    gemm(x_b, IN, W1p,                 H, H1,        1536, 0, b1,          N, H, IN); // t0 -> H1[:,0:512] f32
    gemm(x_b, IN, W1p + (IN/2)*H,      H, T1p,       H,    2, b1 + H,      N, H, IN); // t1 (packed)
    gemm(x_b, IN, W1p + 2*(IN/2)*H,    H, T2p,       H,    2, b1 + 2*H,    N, H, IN); // t2 (packed)
    gemm(adj_b, N, T1p, H, H1 + 512,  1536, 0, zbias, N, H, N);   // adj@t1   -> H1[:,512:1024]
    gemm(adj_b, N, T2p, H, T1p,       H,    2, zbias, N, H, N);   // adj@t2   -> T1p (packed, reuse)
    gemm(adj_b, N, T1p, H, H1 + 1024, 1536, 0, zbias, N, H, N);   // adj^2@t2 -> H1[:,1024:1536]

    // ---- BatchNorm + ReLU -> bf16 ----
    col_partial_stats<<<dim3(1536 / 256, N / 32), 256, 0, stream>>>(H1, 1536, 32, sums, sqs);
    bn_relu_to_bf16<<<dim3((N * 1536 + 255) / 256), 256, 0, stream>>>(
        H1, H1b, N * 1536, 1536, 1.0f / (float)N, sums, sqs, gamma, beta);

    // ---- Layer 2 ----
    gemm(H1b, 1536, W2p,                    OUT, H2b,       768, 1, b2,         N, OUT, 1536); // u0 -> H2b[:,0:256]
    gemm(H1b, 1536, W2p + (1536/2)*OUT,     OUT, U1p,       OUT, 2, b2 + OUT,   N, OUT, 1536); // u1 (packed)
    gemm(H1b, 1536, W2p + 2*(1536/2)*OUT,   OUT, U2p,       OUT, 2, b2 + 2*OUT, N, OUT, 1536); // u2 (packed)
    gemm(adj_b, N, U1p, OUT, H2b + 256, 768, 1, zbias, N, OUT, N);   // adj@u1   -> H2b[:,256:512]
    gemm(adj_b, N, U2p, OUT, Vp,        OUT, 2, zbias, N, OUT, N);   // adj@u2   (packed)
    gemm(adj_b, N, Vp,  OUT, H2b + 512, 768, 1, zbias, N, OUT, N);   // adj^2@u2 -> H2b[:,512:768]

    // ---- Final projection ----
    gemm(H2b, 768, Wfp, OUT, out, OUT, 0, bf, N, OUT, 768);
}
